// DecoderBlock_2465311228638
// MI455X (gfx1250) — matrix operand order repaired and checked by reading, not compiled, not run on hardware
//
#include <hip/hip_runtime.h>

// ---------------------------------------------------------------------------
// Decoder block for MI455X (gfx1250, wave32, WMMA).
// Pipeline: convert/transpose weights fp32->bf16 | RMSNorm | QKV GEMMs |
// RoPE | flash attention (WMMA) | Wo GEMM + residual | LayerNorm |
// FFN GEMMs with fused SiLU-gate | W2 GEMM + residual -> d_out (fp32).
// Matrix math: V_WMMA_F32_16X16X32_BF16. GEMM staging: async global->LDS DMA
// (GLOBAL_LOAD_ASYNC_TO_LDS_B128 + s_wait_asynccnt), double-buffered.
// ---------------------------------------------------------------------------

#define HEADS 16
#define DIMSZ 2048
#define DFFSZ 4096
#define HDSZ  128            // DIMSZ / HEADS
#define BATCH 2
#define SEQ   2048
#define MTOT  (BATCH * SEQ)  // 4096 rows for all GEMMs
#define ROPE_THETA 10000.0f
#define RMS_EPS 1.1920928955078125e-07f
#define LN_EPS  1e-5f
#define NEG_INF (-__builtin_inff())

typedef unsigned short u16;
typedef unsigned int   u32;
typedef __attribute__((ext_vector_type(16))) __bf16 v16bf;
typedef __attribute__((ext_vector_type(8)))  float  v8f;

struct __align__(16) B128 { u32 w0, w1, w2, w3; };   // POD 16-byte vector

union Frag16 { v16bf v; B128 q[2]; u16 u[16]; };     // 16x32 bf16 operand frag
union Acc8   { v8f   v; float f[8]; };               // 16x16 f32 accumulator

__device__ __forceinline__ u16 f2bf(float f) {
    u32 u = __float_as_uint(f);
    u32 r = u + 0x7FFFu + ((u >> 16) & 1u);          // round-to-nearest-even
    return (u16)(r >> 16);
}
__device__ __forceinline__ float bf2f(u16 h) { return __uint_as_float(((u32)h) << 16); }

__device__ __forceinline__ v8f wmma_bf16(const Frag16& a, const Frag16& b, v8f c) {
    return __builtin_amdgcn_wmma_f32_16x16x32_bf16(
        false, a.v, false, b.v, (short)0, c, false, false);
}

// addrspace(3) offset == low 32 bits of the generic pointer (LDS aperture)
__device__ __forceinline__ u32 lds_addr(const void* p) {
    return (u32)(uintptr_t)p;
}

// ---------------------------------------------------------------------------
// Weight convert + transpose: W (K x N, f32) -> Wt (N x K, bf16)
// ---------------------------------------------------------------------------
__global__ __launch_bounds__(256)
void convert_transpose(const float* __restrict__ W, u16* __restrict__ Wt, int K, int N) {
    __shared__ float tile[32][33];
    const int nb = blockIdx.x * 32, kb = blockIdx.y * 32;
    const int tx = threadIdx.x & 31, ty = threadIdx.x >> 5;
#pragma unroll
    for (int it = 0; it < 4; ++it)
        tile[ty + 8 * it][tx] = W[(size_t)(kb + ty + 8 * it) * N + nb + tx];
    __syncthreads();
#pragma unroll
    for (int it = 0; it < 4; ++it)
        Wt[(size_t)(nb + ty + 8 * it) * K + kb + tx] = f2bf(tile[tx][ty + 8 * it]);
}

// ---------------------------------------------------------------------------
// Row-wise RMSNorm -> bf16
// ---------------------------------------------------------------------------
__global__ __launch_bounds__(256)
void rmsnorm_kernel(const float* __restrict__ x, const float* __restrict__ g,
                    u16* __restrict__ h) {
    __shared__ float red[8];
    const int row = blockIdx.x;
    const float* xr = x + (size_t)row * DIMSZ;
    float ss = 0.f;
    for (int i = threadIdx.x; i < DIMSZ; i += 256) { float v = xr[i]; ss += v * v; }
#pragma unroll
    for (int m = 16; m >= 1; m >>= 1) ss += __shfl_xor(ss, m);
    if ((threadIdx.x & 31) == 0) red[threadIdx.x >> 5] = ss;
    __syncthreads();
    if (threadIdx.x < 8) {
        float v = red[threadIdx.x];
#pragma unroll
        for (int m = 4; m >= 1; m >>= 1) v += __shfl_xor(v, m, 8);
        if (threadIdx.x == 0) red[0] = v;
    }
    __syncthreads();
    const float r = rsqrtf(red[0] / DIMSZ + RMS_EPS);
    for (int i = threadIdx.x; i < DIMSZ; i += 256)
        h[(size_t)row * DIMSZ + i] = f2bf(xr[i] * r * g[i]);
}

// ---------------------------------------------------------------------------
// Row-wise LayerNorm -> bf16
// ---------------------------------------------------------------------------
__global__ __launch_bounds__(256)
void layernorm_kernel(const float* __restrict__ x, const float* __restrict__ g,
                      const float* __restrict__ b, u16* __restrict__ h) {
    __shared__ float redS[8], redQ[8];
    const int row = blockIdx.x;
    const float* xr = x + (size_t)row * DIMSZ;
    float s1 = 0.f, s2 = 0.f;
    for (int i = threadIdx.x; i < DIMSZ; i += 256) { float v = xr[i]; s1 += v; s2 += v * v; }
#pragma unroll
    for (int m = 16; m >= 1; m >>= 1) { s1 += __shfl_xor(s1, m); s2 += __shfl_xor(s2, m); }
    if ((threadIdx.x & 31) == 0) { redS[threadIdx.x >> 5] = s1; redQ[threadIdx.x >> 5] = s2; }
    __syncthreads();
    if (threadIdx.x < 8) {
        float a = redS[threadIdx.x], c = redQ[threadIdx.x];
#pragma unroll
        for (int m = 4; m >= 1; m >>= 1) { a += __shfl_xor(a, m, 8); c += __shfl_xor(c, m, 8); }
        if (threadIdx.x == 0) { redS[0] = a; redQ[0] = c; }
    }
    __syncthreads();
    const float mu  = redS[0] / DIMSZ;
    const float var = redQ[0] / DIMSZ - mu * mu;
    const float r   = rsqrtf(var + LN_EPS);
    for (int i = threadIdx.x; i < DIMSZ; i += 256)
        h[(size_t)row * DIMSZ + i] = f2bf((xr[i] - mu) * r * g[i] + b[i]);
}

// ---------------------------------------------------------------------------
// RoPE, in place on bf16 (B,T,DIM): pairs (2i, 2i+1) within each head
// ---------------------------------------------------------------------------
__global__ __launch_bounds__(256)
void rope_kernel(u16* __restrict__ qk, int total_pairs) {
    const int p = blockIdx.x * 256 + threadIdx.x;
    if (p >= total_pairs) return;
    const int d2 = p % (DIMSZ / 2);
    const int t  = (p / (DIMSZ / 2)) % SEQ;
    const int j  = d2 % (HDSZ / 2);
    const float inv = __expf(-(2.0f * (float)j / (float)HDSZ) * __logf(ROPE_THETA));
    const float ang = (float)t * inv;
    float s, c; __sincosf(ang, &s, &c);
    const float x1 = bf2f(qk[2 * p]), x2 = bf2f(qk[2 * p + 1]);
    qk[2 * p]     = f2bf(x1 * c - x2 * s);
    qk[2 * p + 1] = f2bf(x1 * s + x2 * c);
}

// ---------------------------------------------------------------------------
// Tiled WMMA GEMM: C[M,N] = A[M,K](bf16,row) @ Bt[N,K](bf16,row)^T (+bias)
// 128x128x32 block tile, 8 waves (4x2), 32x64 per wave.
// Tiles staged by async global->LDS DMA, double-buffered, 1 barrier/k-step.
// ---------------------------------------------------------------------------
#define BM 128
#define BN 128
#define BK 32
#define LDK 40   // u16 stride (32 + 8 pad), keeps 16B alignment

enum { EPI_BF16 = 0, EPI_BF16_SILU = 1, EPI_BF16_MUL = 2, EPI_F32_RESID = 3 };

__global__ __launch_bounds__(256, 2)
void gemm_bf16_wmma(const u16* __restrict__ A, const u16* __restrict__ Bt,
                    const float* __restrict__ bias,
                    u16* __restrict__ outB, float* __restrict__ outF,
                    const u16* __restrict__ mulB, const float* __restrict__ resid,
                    int M, int N, int K, int mode) {
    __shared__ __align__(16) u16 ldsA[2][BM * LDK];
    __shared__ __align__(16) u16 ldsB[2][BN * LDK];

    const int tid  = threadIdx.x;
    const int lane = tid & 31;
    const int wave = tid >> 5;
    const int wr = wave >> 1, wc = wave & 1;
    const int laneRow = lane & 15, laneHi = lane >> 4;
    const int mBase = blockIdx.y * BM, nBase = blockIdx.x * BN;

    // cooperative tile mapping: each thread owns a 32B slice of A and B tiles
    const int lrow  = tid >> 1;
    const int lhalf = (tid & 1) * 16;
    const u16* aPtr = A  + (size_t)(mBase + lrow) * K + lhalf;
    const u16* bPtr = Bt + (size_t)(nBase + lrow) * K + lhalf;
    const u32 ldsAo[2] = { lds_addr(&ldsA[0][lrow * LDK + lhalf]),
                           lds_addr(&ldsA[1][lrow * LDK + lhalf]) };
    const u32 ldsBo[2] = { lds_addr(&ldsB[0][lrow * LDK + lhalf]),
                           lds_addr(&ldsB[1][lrow * LDK + lhalf]) };

    // async DMA: INST_OFFSET applies to both global and LDS addresses, so one
    // VGPR pair + offset:16 covers this thread's two 16B chunks on both sides.
    auto issue_tile = [&](int kb, int buf) {
        const u16* ap = aPtr + (size_t)kb * BK;
        const u16* bp = bPtr + (size_t)kb * BK;
        asm volatile(
            "global_load_async_to_lds_b128 %0, %2, off\n\t"
            "global_load_async_to_lds_b128 %0, %2, off offset:16\n\t"
            "global_load_async_to_lds_b128 %1, %3, off\n\t"
            "global_load_async_to_lds_b128 %1, %3, off offset:16"
            :: "v"(ldsAo[buf]), "v"(ldsBo[buf]), "v"(ap), "v"(bp)
            : "memory");
    };

    Acc8 acc[2][4];
#pragma unroll
    for (int i = 0; i < 2; ++i)
#pragma unroll
        for (int j = 0; j < 4; ++j)
#pragma unroll
            for (int e = 0; e < 8; ++e) acc[i][j].f[e] = 0.f;

    const int nk = K / BK;
    issue_tile(0, 0);
    for (int kb = 0; kb < nk; ++kb) {
        const int buf = kb & 1;
        asm volatile("s_wait_asynccnt 0x0" ::: "memory");
        __syncthreads();
        if (kb + 1 < nk) {
            issue_tile(kb + 1, buf ^ 1);   // DMA overlaps with WMMA burst
            if (kb + 2 < nk) {             // pull tile after next into GL2
                __builtin_prefetch(aPtr + (size_t)(kb + 2) * BK, 0, 0);
                __builtin_prefetch(bPtr + (size_t)(kb + 2) * BK, 0, 0);
            }
        }
        Frag16 af[2], bf[4];
#pragma unroll
        for (int i = 0; i < 2; ++i) {
            const u16* p = &ldsA[buf][(wr * 32 + i * 16 + laneRow) * LDK + laneHi * 8];
            af[i].q[0] = *(const B128*)(p);
            af[i].q[1] = *(const B128*)(p + 16);
        }
#pragma unroll
        for (int j = 0; j < 4; ++j) {
            const u16* p = &ldsB[buf][(wc * 64 + j * 16 + laneRow) * LDK + laneHi * 8];
            bf[j].q[0] = *(const B128*)(p);
            bf[j].q[1] = *(const B128*)(p + 16);
        }
#pragma unroll
        for (int i = 0; i < 2; ++i)
#pragma unroll
            for (int j = 0; j < 4; ++j)
                acc[i][j].v = wmma_bf16(af[i], bf[j], acc[i][j].v);
    }

    // epilogue
#pragma unroll
    for (int i = 0; i < 2; ++i) {
#pragma unroll
        for (int j = 0; j < 4; ++j) {
            const int col = nBase + wc * 64 + j * 16 + laneRow;
            const float bv = bias ? bias[col] : 0.0f;
#pragma unroll
            for (int e = 0; e < 8; ++e) {
                const int row = mBase + wr * 32 + i * 16 + e + laneHi * 8;
                const size_t idx = (size_t)row * N + col;
                float v = acc[i][j].f[e] + bv;
                if (mode == EPI_BF16) {
                    outB[idx] = f2bf(v);
                } else if (mode == EPI_BF16_SILU) {
                    outB[idx] = f2bf(v / (1.0f + __expf(-v)));
                } else if (mode == EPI_BF16_MUL) {
                    outB[idx] = f2bf(v * bf2f(mulB[idx]));
                } else {
                    outF[idx] = resid[idx] + v;
                }
            }
        }
    }
}

// ---------------------------------------------------------------------------
// Flash attention (causal + padding), WMMA for QK^T and PV.
// Block: (b, h, 128 q rows); 8 waves x 16 q rows; 32-key blocks via LDS.
// ---------------------------------------------------------------------------
#define BQ  128
#define BKV 32
#define KLD 136   // ldsK row stride (dims + pad)
#define VLD 40    // ldsVt row stride (keys + pad)
#define PLD 40

__global__ __launch_bounds__(256, 1)
void attention_kernel(const u16* __restrict__ qg, const u16* __restrict__ kg,
                      const u16* __restrict__ vg, const int* __restrict__ pad,
                      u16* __restrict__ outg) {
    __shared__ __align__(16) u16 ldsK[BKV * KLD];
    __shared__ __align__(16) u16 ldsVt[HDSZ * VLD];
    __shared__ __align__(16) u16 ldsP[8][16 * PLD];
    __shared__ int ldsPad[BKV];

    const int tid = threadIdx.x, lane = tid & 31, wave = tid >> 5;
    const int laneRow = lane & 15, laneHi = lane >> 4;
    const int b = blockIdx.z, h = blockIdx.y, qb = blockIdx.x * BQ;
    const int qRow0 = qb + wave * 16;

    const u16* qbase = qg + (size_t)(b * SEQ) * DIMSZ + h * HDSZ;
    const u16* kbase = kg + (size_t)(b * SEQ) * DIMSZ + h * HDSZ;
    const u16* vbase = vg + (size_t)(b * SEQ) * DIMSZ + h * HDSZ;

    // resident Q fragments (4 k-chunks of 32 dims), loaded directly from global
    Frag16 qf[4];
    {
        const u16* qp = qbase + (size_t)(qRow0 + laneRow) * DIMSZ + laneHi * 8;
#pragma unroll
        for (int kk = 0; kk < 4; ++kk) {
            qf[kk].q[0] = *(const B128*)(qp + kk * 32);
            qf[kk].q[1] = *(const B128*)(qp + kk * 32 + 16);
        }
    }

    Acc8 o[8];
#pragma unroll
    for (int n = 0; n < 8; ++n)
#pragma unroll
        for (int e = 0; e < 8; ++e) o[n].f[e] = 0.f;
    float mrow[8], lrow[8];
#pragma unroll
    for (int e = 0; e < 8; ++e) { mrow[e] = NEG_INF; lrow[e] = 0.f; }

    const float scl = 0.08838834764831845f;  // 1/sqrt(128)
    const int nkb = (qb + BQ + BKV - 1) / BKV;  // causal key-block bound

    for (int kb = 0; kb < nkb; ++kb) {
        const int key0 = kb * BKV;
        {   // cooperative K / V^T / pad loads
            const int krow = tid >> 3;
            const int part = (tid & 7) * 16;
            const u16* kp = kbase + (size_t)(key0 + krow) * DIMSZ + part;
            *(B128*)&ldsK[krow * KLD + part]     = *(const B128*)(kp);
            *(B128*)&ldsK[krow * KLD + part + 8] = *(const B128*)(kp + 8);
            const u16* vp = vbase + (size_t)(key0 + krow) * DIMSZ + part;
            union { B128 b; u16 u[8]; } v0, v1;
            v0.b = *(const B128*)(vp); v1.b = *(const B128*)(vp + 8);
#pragma unroll
            for (int e = 0; e < 8; ++e) ldsVt[(part + e) * VLD + krow]     = v0.u[e];
#pragma unroll
            for (int e = 0; e < 8; ++e) ldsVt[(part + 8 + e) * VLD + krow] = v1.u[e];
            if (tid < BKV) ldsPad[tid] = pad[b * SEQ + key0 + tid];
        }
        __syncthreads();

        // S = Q @ K^T  (two 16x16 tiles: keys 0-15 and 16-31)
        Acc8 s0, s1;
#pragma unroll
        for (int e = 0; e < 8; ++e) { s0.f[e] = 0.f; s1.f[e] = 0.f; }
#pragma unroll
        for (int kk = 0; kk < 4; ++kk) {
            Frag16 b0, b1;
            const u16* p0 = &ldsK[laneRow * KLD + kk * 32 + laneHi * 8];
            b0.q[0] = *(const B128*)(p0); b0.q[1] = *(const B128*)(p0 + 16);
            const u16* p1 = &ldsK[(16 + laneRow) * KLD + kk * 32 + laneHi * 8];
            b1.q[0] = *(const B128*)(p1); b1.q[1] = *(const B128*)(p1 + 16);
            s0.v = wmma_bf16(qf[kk], b0, s0.v);
            s1.v = wmma_bf16(qf[kk], b1, s1.v);
        }

        // masking + online softmax (row spread across 16 lanes -> shfl_xor)
        const int keyA = key0 + laneRow, keyB = keyA + 16;
        const int okA = ldsPad[laneRow] != 0, okB = ldsPad[laneRow + 16] != 0;
        float pA[8], pB[8], alpha[8];
#pragma unroll
        for (int e = 0; e < 8; ++e) {
            const int row = qRow0 + e + laneHi * 8;
            float vA = (okA && keyA <= row) ? s0.f[e] * scl : NEG_INF;
            float vB = (okB && keyB <= row) ? s1.f[e] * scl : NEG_INF;
            float t = fmaxf(vA, vB);
            t = fmaxf(t, __shfl_xor(t, 1)); t = fmaxf(t, __shfl_xor(t, 2));
            t = fmaxf(t, __shfl_xor(t, 4)); t = fmaxf(t, __shfl_xor(t, 8));
            const float mnew = fmaxf(mrow[e], t);
            const float a = (mrow[e] == NEG_INF) ? 0.0f : __expf(mrow[e] - mnew);
            pA[e] = (vA == NEG_INF) ? 0.0f : __expf(vA - mnew);
            pB[e] = (vB == NEG_INF) ? 0.0f : __expf(vB - mnew);
            float rs = pA[e] + pB[e];
            rs += __shfl_xor(rs, 1); rs += __shfl_xor(rs, 2);
            rs += __shfl_xor(rs, 4); rs += __shfl_xor(rs, 8);
            lrow[e] = lrow[e] * a + rs;
            mrow[e] = mnew;
            alpha[e] = a;
        }
#pragma unroll
        for (int n = 0; n < 8; ++n)
#pragma unroll
            for (int e = 0; e < 8; ++e) o[n].f[e] *= alpha[e];

        // C-layout P -> A-frag layout via wave-private LDS patch
#pragma unroll
        for (int e = 0; e < 8; ++e) {
            const int r = e + laneHi * 8;
            ldsP[wave][r * PLD + laneRow]      = f2bf(pA[e]);
            ldsP[wave][r * PLD + 16 + laneRow] = f2bf(pB[e]);
        }
        Frag16 pf;
        {
            const u16* pp = &ldsP[wave][laneRow * PLD + laneHi * 8];
            pf.q[0] = *(const B128*)(pp);
            pf.q[1] = *(const B128*)(pp + 16);
        }
        // O += P @ V  (8 dim-tiles of 16)
#pragma unroll
        for (int n = 0; n < 8; ++n) {
            Frag16 vf;
            const u16* vp2 = &ldsVt[(n * 16 + laneRow) * VLD + laneHi * 8];
            vf.q[0] = *(const B128*)(vp2); vf.q[1] = *(const B128*)(vp2 + 16);
            o[n].v = wmma_bf16(pf, vf, o[n].v);
        }
        __syncthreads();
    }

    // normalize + write bf16 (b, t, h*HD + d)
    u16* ob = outg + (size_t)(b * SEQ) * DIMSZ + h * HDSZ;
#pragma unroll
    for (int e = 0; e < 8; ++e) {
        const int row = qRow0 + e + laneHi * 8;
        const float inv = (lrow[e] > 0.f) ? 1.0f / lrow[e] : 0.0f;
#pragma unroll
        for (int n = 0; n < 8; ++n)
            ob[(size_t)row * DIMSZ + n * 16 + laneRow] = f2bf(o[n].f[e] * inv);
    }
}

// ---------------------------------------------------------------------------
// Host-side orchestration
// ---------------------------------------------------------------------------
extern "C" void kernel_launch(void* const* d_in, const int* in_sizes, int n_in,
                              void* d_out, int out_size, void* d_ws, size_t ws_size,
                              hipStream_t stream) {
    (void)in_sizes; (void)n_in; (void)out_size; (void)ws_size;
    const float* x    = (const float*)d_in[0];
    const float* Wq   = (const float*)d_in[1];
    const float* bq   = (const float*)d_in[2];
    const float* Wk   = (const float*)d_in[3];
    const float* bk   = (const float*)d_in[4];
    const float* Wv   = (const float*)d_in[5];
    const float* Wo   = (const float*)d_in[6];
    const float* bo   = (const float*)d_in[7];
    const float* W1   = (const float*)d_in[8];
    const float* b1   = (const float*)d_in[9];
    const float* Wg1  = (const float*)d_in[10];
    const float* bg1  = (const float*)d_in[11];
    const float* Wg2  = (const float*)d_in[12];
    const float* bg2  = (const float*)d_in[13];
    const float* W2   = (const float*)d_in[14];
    const float* b2   = (const float*)d_in[15];
    const float* grms = (const float*)d_in[16];
    const float* gln  = (const float*)d_in[17];
    const float* bln  = (const float*)d_in[18];
    const int*   pad  = (const int*)d_in[19];

    char* w = (char*)d_ws;
    auto alloc = [&](size_t bytes) -> char* {
        char* p = w; w += (bytes + 255) & ~(size_t)255; return p;
    };
    u16* WqT = (u16*)alloc((size_t)DIMSZ * DIMSZ * 2);
    u16* WkT = (u16*)alloc((size_t)DIMSZ * DIMSZ * 2);
    u16* WvT = (u16*)alloc((size_t)DIMSZ * DIMSZ * 2);
    u16* WoT = (u16*)alloc((size_t)DIMSZ * DIMSZ * 2);
    u16* W1T = (u16*)alloc((size_t)DIMSZ * DFFSZ * 2);
    u16* Wg1T = (u16*)alloc((size_t)DFFSZ * DFFSZ * 2);
    u16* Wg2T = (u16*)alloc((size_t)DFFSZ * DFFSZ * 2);
    u16* W2T = (u16*)alloc((size_t)DFFSZ * DIMSZ * 2);
    u16* hbf  = (u16*)alloc((size_t)MTOT * DIMSZ * 2);
    u16* qbf  = (u16*)alloc((size_t)MTOT * DIMSZ * 2);
    u16* kbf  = (u16*)alloc((size_t)MTOT * DIMSZ * 2);
    u16* vbf  = (u16*)alloc((size_t)MTOT * DIMSZ * 2);
    u16* abf  = (u16*)alloc((size_t)MTOT * DIMSZ * 2);
    float* x1 = (float*)alloc((size_t)MTOT * DIMSZ * 4);
    u16* h2bf = (u16*)alloc((size_t)MTOT * DIMSZ * 2);
    u16* ubf  = (u16*)alloc((size_t)MTOT * DFFSZ * 2);
    u16* t1bf = (u16*)alloc((size_t)MTOT * DFFSZ * 2);
    u16* sbf  = (u16*)alloc((size_t)MTOT * DFFSZ * 2);

    // 1. weights -> bf16 transposed (N x K)
    convert_transpose<<<dim3(DIMSZ/32, DIMSZ/32), 256, 0, stream>>>(Wq, WqT, DIMSZ, DIMSZ);
    convert_transpose<<<dim3(DIMSZ/32, DIMSZ/32), 256, 0, stream>>>(Wk, WkT, DIMSZ, DIMSZ);
    convert_transpose<<<dim3(DIMSZ/32, DIMSZ/32), 256, 0, stream>>>(Wv, WvT, DIMSZ, DIMSZ);
    convert_transpose<<<dim3(DIMSZ/32, DIMSZ/32), 256, 0, stream>>>(Wo, WoT, DIMSZ, DIMSZ);
    convert_transpose<<<dim3(DFFSZ/32, DIMSZ/32), 256, 0, stream>>>(W1, W1T, DIMSZ, DFFSZ);
    convert_transpose<<<dim3(DFFSZ/32, DFFSZ/32), 256, 0, stream>>>(Wg1, Wg1T, DFFSZ, DFFSZ);
    convert_transpose<<<dim3(DFFSZ/32, DFFSZ/32), 256, 0, stream>>>(Wg2, Wg2T, DFFSZ, DFFSZ);
    convert_transpose<<<dim3(DIMSZ/32, DFFSZ/32), 256, 0, stream>>>(W2, W2T, DFFSZ, DIMSZ);

    // 2. RMSNorm
    rmsnorm_kernel<<<MTOT, 256, 0, stream>>>(x, grms, hbf);

    // 3. QKV projections
    gemm_bf16_wmma<<<dim3(DIMSZ/BN, MTOT/BM), 256, 0, stream>>>(
        hbf, WqT, bq, qbf, nullptr, nullptr, nullptr, MTOT, DIMSZ, DIMSZ, EPI_BF16);
    gemm_bf16_wmma<<<dim3(DIMSZ/BN, MTOT/BM), 256, 0, stream>>>(
        hbf, WkT, bk, kbf, nullptr, nullptr, nullptr, MTOT, DIMSZ, DIMSZ, EPI_BF16);
    gemm_bf16_wmma<<<dim3(DIMSZ/BN, MTOT/BM), 256, 0, stream>>>(
        hbf, WvT, nullptr, vbf, nullptr, nullptr, nullptr, MTOT, DIMSZ, DIMSZ, EPI_BF16);

    // 4. RoPE on q and k
    const int pairs = MTOT * DIMSZ / 2;
    rope_kernel<<<(pairs + 255) / 256, 256, 0, stream>>>(qbf, pairs);
    rope_kernel<<<(pairs + 255) / 256, 256, 0, stream>>>(kbf, pairs);

    // 5. attention
    attention_kernel<<<dim3(SEQ/BQ, HEADS, BATCH), 256, 0, stream>>>(qbf, kbf, vbf, pad, abf);

    // 6. x1 = x + attn @ Wo + bo
    gemm_bf16_wmma<<<dim3(DIMSZ/BN, MTOT/BM), 256, 0, stream>>>(
        abf, WoT, bo, nullptr, x1, nullptr, x, MTOT, DIMSZ, DIMSZ, EPI_F32_RESID);

    // 7. LayerNorm
    layernorm_kernel<<<MTOT, 256, 0, stream>>>(x1, gln, bln, h2bf);

    // 8. u = h2 @ W1 + b1
    gemm_bf16_wmma<<<dim3(DFFSZ/BN, MTOT/BM), 256, 0, stream>>>(
        h2bf, W1T, b1, ubf, nullptr, nullptr, nullptr, MTOT, DFFSZ, DIMSZ, EPI_BF16);

    // 9. t1 = silu(u @ Wg1 + bg1)
    gemm_bf16_wmma<<<dim3(DFFSZ/BN, MTOT/BM), 256, 0, stream>>>(
        ubf, Wg1T, bg1, t1bf, nullptr, nullptr, nullptr, MTOT, DFFSZ, DFFSZ, EPI_BF16_SILU);

    // 10. s = t1 * (u @ Wg2 + bg2)
    gemm_bf16_wmma<<<dim3(DFFSZ/BN, MTOT/BM), 256, 0, stream>>>(
        ubf, Wg2T, bg2, sbf, nullptr, t1bf, nullptr, MTOT, DFFSZ, DFFSZ, EPI_BF16_MUL);

    // 11. out = x1 + s @ W2 + b2
    gemm_bf16_wmma<<<dim3(DIMSZ/BN, MTOT/BM), 256, 0, stream>>>(
        sbf, W2T, b2, nullptr, (float*)d_out, nullptr, x1, MTOT, DIMSZ, DFFSZ, EPI_F32_RESID);
}
